// VSSBlock3D_46291157517007
// MI455X (gfx1250) — matrix operand order repaired and checked by reading, not compiled, not run on hardware
//
#include <hip/hip_runtime.h>
#include <hip/hip_bf16.h>
#include <math.h>

// ---- problem constants (B=1) ----
#define LTOT 8192
#define CDIM 96
#define DINC 96
#define KDIR 8
#define NST  16
#define RRK  6
#define EDIM 38      // R + 2N
#define DD   16
#define HH   16
#define WW   32

typedef __attribute__((ext_vector_type(16))) _Float16 v16h;
typedef __attribute__((ext_vector_type(8)))  _Float16 v8h;
typedef __attribute__((ext_vector_type(8)))  float    v8f;

__device__ __forceinline__ float siluf(float x){ return x / (1.0f + expf(-x)); }

// ---- WMMA fragment loaders (16-bit A 16x32, B 32x16 layouts per CDNA5 ISA) ----
// A: lane<16 -> row m, K {8h*8..+7} and {16+8h*8..}; contiguous 16B chunks.
__device__ __forceinline__ v16h load_a_frag(const _Float16* rowbase, int kbase, bool valid){
  int lane = threadIdx.x & 31; int half = lane >> 4;
  const _Float16* p = rowbase + kbase + half*8;
  v8h lo = {}, hi = {};
  if (valid){ lo = *(const v8h*)(p); hi = *(const v8h*)(p + 16); }
  return __builtin_shufflevector(lo, hi, 0,1,2,3,4,5,6,7,8,9,10,11,12,13,14,15);
}
// B: lane<16 -> col n, K {half*8 .. +7} and {16+half*8 .. +7}; two contiguous 16B chunks.
__device__ __forceinline__ v16h load_b_frag(const _Float16* colbase, int kbase, bool valid){
  int lane = threadIdx.x & 31; int half = lane >> 4;
  const _Float16* p = colbase + kbase + half*8;
  v8h lo = {}, hi = {};
  if (valid){ lo = *(const v8h*)(p); hi = *(const v8h*)(p + 16); }
  return __builtin_shufflevector(lo, hi, 0,1,2,3,4,5,6,7,8,9,10,11,12,13,14,15);
}
__device__ __forceinline__ v8f wmma_f16(v16h a, v16h b, v8f c){
  return __builtin_amdgcn_wmma_f32_16x16x32_f16(false, a, false, b, (short)0, c, false, false);
}

// ---- weight conversion ----
__global__ void k_f2h(const float* __restrict__ src, _Float16* __restrict__ dst, int n){
  int i = blockIdx.x*blockDim.x + threadIdx.x;
  if (i < n) dst[i] = (_Float16)src[i];
}
// cab_w1 (32,96,27) -> [o][tap][i] f16
__global__ void k_tw1(const float* __restrict__ w, _Float16* __restrict__ wt){
  int i = blockIdx.x*blockDim.x + threadIdx.x;
  if (i >= 32*27*96) return;
  int o = i/(27*96), r = i%(27*96), tap = r/96, c = r%96;
  wt[i] = (_Float16)w[(o*96 + c)*27 + tap];
}
// cab_w2 (96,32,27) -> [o][tap][i] f16
__global__ void k_tw2(const float* __restrict__ w, _Float16* __restrict__ wt){
  int i = blockIdx.x*blockDim.x + threadIdx.x;
  if (i >= 96*27*32) return;
  int o = i/(27*32), r = i%(27*32), tap = r/32, c = r%32;
  wt[i] = (_Float16)w[(o*32 + c)*27 + tap];
}

// ---- scan-position -> voxel map for the 8 directions ----
__global__ void k_vmap(int* __restrict__ vmap){
  int idx = blockIdx.x*blockDim.x + threadIdx.x;
  if (idx >= KDIR*LTOT) return;
  int k = idx / LTOT, l = idx % LTOT;
  int kk = k & 3;
  int pos = (k < 4) ? l : (LTOT - 1 - l);
  int p0,p1,p2;
  if      (kk==0){p0=0;p1=1;p2=2;}
  else if (kk==1){p0=1;p1=2;p2=0;}
  else if (kk==2){p0=2;p1=0;p2=1;}
  else           {p0=2;p1=1;p2=0;}
  int S[3] = {DD,HH,WW};
  int d2 = S[p2], d1 = S[p1];
  int i0 = pos/(d1*d2); int r2 = pos%(d1*d2); int i1 = r2/d2; int i2 = r2%d2;
  int coord[3]; coord[p0]=i0; coord[p1]=i1; coord[p2]=i2;
  vmap[idx] = (coord[0]*HH + coord[1])*WW + coord[2];
}

// ---- LayerNorm1 over C, emit f16 ----
__global__ void k_ln1(const float* __restrict__ inp, const float* __restrict__ w,
                      const float* __restrict__ b, _Float16* __restrict__ xn_h){
  int l = blockIdx.x*blockDim.x + threadIdx.x;
  if (l >= LTOT) return;
  const float* row = inp + l*CDIM;
  float m = 0.f;
  for (int c=0;c<CDIM;++c) m += row[c];
  m /= (float)CDIM;
  float v = 0.f;
  for (int c=0;c<CDIM;++c){ float d = row[c]-m; v += d*d; }
  v /= (float)CDIM;
  float r = rsqrtf(v + 1e-6f);
  for (int c=0;c<CDIM;++c) xn_h[l*CDIM + c] = (_Float16)((row[c]-m)*r*w[c] + b[c]);
}

// ---- in_proj GEMM: (L x 96) x (96 x 192); 1 wave -> 16x64 block (4 N-tiles) ----
__global__ void k_inproj(const _Float16* __restrict__ xn_h, const _Float16* __restrict__ win_h,
                         float* __restrict__ xf, float* __restrict__ zf){
  int l0 = blockIdx.x*16, nb = blockIdx.y*64;
  int lane = threadIdx.x & 31; int half = lane >> 4; int m = lane & 15;
  const _Float16* arow = xn_h + (l0 + m)*CDIM;
  v8f acc[4] = {};
#pragma unroll
  for (int kc=0; kc<3; ++kc){
    v16h a = load_a_frag(arow, kc*32, true);
#pragma unroll
    for (int t=0;t<4;++t){
      v16h b = load_b_frag(win_h + (size_t)(nb + t*16 + m)*CDIM, kc*32, true);
      acc[t] = wmma_f16(a, b, acc[t]);
    }
  }
#pragma unroll
  for (int t=0;t<4;++t){
    int e = nb + t*16 + m;
#pragma unroll
    for (int r=0;r<8;++r){
      int l = l0 + r + 8*half;
      float val = acc[t][r];
      if (e < DINC) xf[l*DINC + e] = val;
      else          zf[l*DINC + (e - DINC)] = val;
    }
  }
}

// ---- depthwise 3x3x3 conv + SiLU; emit channel-major f32 and l-major f16 ----
__global__ void k_dwconv(const float* __restrict__ xf, const float* __restrict__ cw,
                         const float* __restrict__ cb, float* __restrict__ xc_f,
                         _Float16* __restrict__ xc_h){
  int tid = blockIdx.x*blockDim.x + threadIdx.x;
  if (tid >= LTOT*DINC) return;
  int l = tid / DINC, c = tid % DINC;
  int d = l/(HH*WW), rem = l%(HH*WW), h = rem/WW, w = rem%WW;
  const float* wk = cw + c*27;
  float s = 0.f;
#pragma unroll
  for (int dz=-1;dz<=1;++dz)
#pragma unroll
   for (int dy=-1;dy<=1;++dy)
#pragma unroll
    for (int dx=-1;dx<=1;++dx){
      int dd=d+dz, h2=h+dy, w2=w+dx;
      if (dd>=0 && dd<DD && h2>=0 && h2<HH && w2>=0 && w2<WW)
        s += wk[(dz+1)*9+(dy+1)*3+(dx+1)] * xf[((dd*HH+h2)*WW+w2)*DINC + c];
    }
  s += cb[c];
  s = siluf(s);
  xc_f[(size_t)c*LTOT + l] = s;
  xc_h[l*DINC + c] = (_Float16)s;
}

// ---- x_proj GEMM per direction; 1 wave -> all 3 e-tiles of one l-tile ----
__global__ void k_xproj(const _Float16* __restrict__ xc_h, const _Float16* __restrict__ wx_h,
                        const int* __restrict__ vmap, float* __restrict__ xdbl){
  int l0 = blockIdx.x*16, k = blockIdx.y;
  int lane = threadIdx.x & 31; int half = lane >> 4; int m = lane & 15;
  int v = vmap[k*LTOT + l0 + m];
  const _Float16* arow = xc_h + (size_t)v*DINC;
  v8f acc[3] = {};
#pragma unroll
  for (int kc=0; kc<3; ++kc){
    v16h a = load_a_frag(arow, kc*32, true);
#pragma unroll
    for (int t=0;t<3;++t){
      int e = t*16 + m;
      bool bvalid = (e < EDIM);
      v16h b = load_b_frag(wx_h + ((size_t)k*EDIM + (bvalid ? e : 0))*DINC, kc*32, bvalid);
      acc[t] = wmma_f16(a, b, acc[t]);
    }
  }
#pragma unroll
  for (int t=0;t<3;++t){
    int e = t*16 + m;
    if (e < EDIM){
#pragma unroll
      for (int r=0;r<8;++r){
        int l = l0 + r + 8*half;
        xdbl[((size_t)k*EDIM + e)*LTOT + l] = acc[t][r];
      }
    }
  }
}

// ---- selective scan: one 16-lane group per (k,d); n = lane%16; reduce over n ----
__global__ void k_scan(const float* __restrict__ xdbl, const float* __restrict__ xc_f,
                       const int* __restrict__ vmap, const float* __restrict__ dt_w,
                       const float* __restrict__ dt_b, const float* __restrict__ A_logs,
                       const float* __restrict__ Ds, float* __restrict__ acc_buf){
  int pair = blockIdx.x*(blockDim.x >> 4) + (threadIdx.x >> 4);
  int n = threadIdx.x & 15;
  if (pair >= KDIR*DINC) return;
  int k = pair / DINC, dch = pair % DINC;
  float Ac = -expf(A_logs[pair*NST + n]);
  float Dc = Ds[pair];
  float bdt = dt_b[pair];
  const float* wr = dt_w + pair*RRK;
  float w0=wr[0], w1=wr[1], w2=wr[2], w3=wr[3], w4=wr[4], w5=wr[5];
  const float* xb   = xdbl + (size_t)k*EDIM*LTOT;
  const int*   vm   = vmap + k*LTOT;
  const float* xrow = xc_f + (size_t)dch*LTOT;
  float h = 0.f;
  for (int l=0; l<LTOT; ++l){
    float dtv = xb[0*LTOT+l]*w0 + xb[1*LTOT+l]*w1 + xb[2*LTOT+l]*w2
              + xb[3*LTOT+l]*w3 + xb[4*LTOT+l]*w4 + xb[5*LTOT+l]*w5 + bdt;
    dtv = (dtv > 20.f) ? dtv : log1pf(expf(dtv));          // softplus
    int v  = vm[l];
    float u  = xrow[v];
    float Bv = xb[(RRK + n)*LTOT + l];
    float Cv = xb[(RRK + NST + n)*LTOT + l];
    h = expf(dtv*Ac)*h + (dtv*u)*Bv;
    float y = h*Cv;
#pragma unroll
    for (int off=1; off<16; off<<=1) y += __shfl_xor(y, off, 16);
    if (n == 0) atomicAdd(&acc_buf[(size_t)dch*LTOT + v], y + Dc*u);
  }
}

// ---- out_norm (LN over DIN) * silu(z), emit f16 ----
__global__ void k_outnorm(const float* __restrict__ acc_buf, const float* __restrict__ zf,
                          const float* __restrict__ onw, const float* __restrict__ onb,
                          _Float16* __restrict__ y_h){
  int l = blockIdx.x*blockDim.x + threadIdx.x;
  if (l >= LTOT) return;
  float m = 0.f;
  for (int e=0;e<DINC;++e) m += acc_buf[(size_t)e*LTOT + l];
  m /= (float)DINC;
  float var = 0.f;
  for (int e=0;e<DINC;++e){ float d = acc_buf[(size_t)e*LTOT + l] - m; var += d*d; }
  var /= (float)DINC;
  float r = rsqrtf(var + 1e-5f);
  for (int e=0;e<DINC;++e){
    float yv = (acc_buf[(size_t)e*LTOT + l] - m)*r*onw[e] + onb[e];
    yv *= siluf(zf[l*DINC + e]);
    y_h[l*DINC + e] = (_Float16)yv;
  }
}

// ---- out_proj GEMM + skip1 residual; 1 wave -> 16x96 block (6 N-tiles) ----
__global__ void k_outproj(const _Float16* __restrict__ y_h, const _Float16* __restrict__ wout_h,
                          const float* __restrict__ inp, const float* __restrict__ skip1,
                          float* __restrict__ x1){
  int l0 = blockIdx.x*16;
  int lane = threadIdx.x & 31; int half = lane >> 4; int m = lane & 15;
  const _Float16* arow = y_h + (l0 + m)*DINC;
  v8f acc[6] = {};
#pragma unroll
  for (int kc=0; kc<3; ++kc){
    v16h a = load_a_frag(arow, kc*32, true);
#pragma unroll
    for (int t=0;t<6;++t){
      v16h b = load_b_frag(wout_h + (size_t)(t*16 + m)*DINC, kc*32, true);
      acc[t] = wmma_f16(a, b, acc[t]);
    }
  }
#pragma unroll
  for (int t=0;t<6;++t){
    int c = t*16 + m;
#pragma unroll
    for (int r=0;r<8;++r){
      int l = l0 + r + 8*half;
      x1[l*CDIM + c] = inp[l*CDIM + c]*skip1[c] + acc[t][r];
    }
  }
}

// ---- LayerNorm2 over C, emit f16 (CAB input) ----
__global__ void k_ln2(const float* __restrict__ x1, const float* __restrict__ w,
                      const float* __restrict__ b, _Float16* __restrict__ y2_h){
  int l = blockIdx.x*blockDim.x + threadIdx.x;
  if (l >= LTOT) return;
  const float* row = x1 + l*CDIM;
  float m = 0.f;
  for (int c=0;c<CDIM;++c) m += row[c];
  m /= (float)CDIM;
  float v = 0.f;
  for (int c=0;c<CDIM;++c){ float d = row[c]-m; v += d*d; }
  v /= (float)CDIM;
  float r = rsqrtf(v + 1e-5f);
  for (int c=0;c<CDIM;++c) y2_h[l*CDIM + c] = (_Float16)((row[c]-m)*r*w[c] + b[c]);
}

// ---- CAB conv1 (96 -> 32, 3x3x3) implicit-GEMM; 1 wave -> both o-tiles ----
__global__ void k_cab1(const _Float16* __restrict__ y2_h, const _Float16* __restrict__ w1t,
                       const float* __restrict__ b1, _Float16* __restrict__ t1_h){
  int l0 = blockIdx.x*16;
  int lane = threadIdx.x & 31; int half = lane >> 4; int m = lane & 15;
  int l = l0 + m;
  int d = l/(HH*WW), rem = l%(HH*WW), h = rem/WW, w = rem%WW;
  v8f acc[2] = {};
  for (int tap=0; tap<27; ++tap){
    int dz = tap/9 - 1, dy = (tap/3)%3 - 1, dx = tap%3 - 1;
    int dd = d+dz, h2 = h+dy, w2 = w+dx;
    bool avalid = (dd>=0 && dd<DD && h2>=0 && h2<HH && w2>=0 && w2<WW);
    const _Float16* arow = y2_h + (size_t)(avalid ? ((dd*HH+h2)*WW+w2) : 0)*DINC;
#pragma unroll
    for (int kc=0; kc<3; ++kc){
      v16h a = load_a_frag(arow, kc*32, avalid);
#pragma unroll
      for (int t=0;t<2;++t){
        v16h b = load_b_frag(w1t + ((size_t)(t*16 + m)*27 + tap)*DINC, kc*32, true);
        acc[t] = wmma_f16(a, b, acc[t]);
      }
    }
  }
#pragma unroll
  for (int t=0;t<2;++t){
    int o = t*16 + m;
#pragma unroll
    for (int r=0;r<8;++r){
      int ll = l0 + r + 8*half;
      float tt = acc[t][r] + b1[o];
      tt = 0.5f*tt*(1.f + erff(tt*0.70710678f));   // exact GELU
      t1_h[ll*32 + o] = (_Float16)tt;
    }
  }
}

// ---- CAB conv2 (32 -> 96, 3x3x3); 1 wave -> all 6 o-tiles + channel sums ----
__global__ void k_cab2(const _Float16* __restrict__ t1_h, const _Float16* __restrict__ w2t,
                       const float* __restrict__ b2, float* __restrict__ t2,
                       float* __restrict__ chan){
  int l0 = blockIdx.x*16;
  int lane = threadIdx.x & 31; int half = lane >> 4; int m = lane & 15;
  int l = l0 + m;
  int d = l/(HH*WW), rem = l%(HH*WW), h = rem/WW, w = rem%WW;
  v8f acc[6] = {};
  for (int tap=0; tap<27; ++tap){
    int dz = tap/9 - 1, dy = (tap/3)%3 - 1, dx = tap%3 - 1;
    int dd = d+dz, h2 = h+dy, w2c = w+dx;
    bool avalid = (dd>=0 && dd<DD && h2>=0 && h2<HH && w2c>=0 && w2c<WW);
    const _Float16* arow = t1_h + (size_t)(avalid ? ((dd*HH+h2)*WW+w2c) : 0)*32;
    v16h a = load_a_frag(arow, 0, avalid);
#pragma unroll
    for (int t=0;t<6;++t){
      v16h b = load_b_frag(w2t + ((size_t)(t*16 + m)*27 + tap)*32, 0, true);
      acc[t] = wmma_f16(a, b, acc[t]);
    }
  }
#pragma unroll
  for (int t=0;t<6;++t){
    int o = t*16 + m;
    float part = 0.f;
#pragma unroll
    for (int r=0;r<8;++r){
      int ll = l0 + r + 8*half;
      float tt = acc[t][r] + b2[o];
      t2[ll*CDIM + o] = tt;
      part += tt;
    }
    atomicAdd(&chan[o], part);
  }
}

// ---- channel attention MLP: mean -> relu fc -> sigmoid fc ----
__global__ void k_ca(const float* __restrict__ chan, const float* __restrict__ w1,
                     const float* __restrict__ bb1, const float* __restrict__ w2,
                     const float* __restrict__ bb2, float* __restrict__ vbuf){
  __shared__ float mS[CDIM];
  __shared__ float v1S[48];
  int t = threadIdx.x;
  if (t < CDIM) mS[t] = chan[t] / (float)LTOT;
  __syncthreads();
  if (t < 48){
    float s = bb1[t];
    for (int c=0;c<CDIM;++c) s += w1[t*CDIM + c]*mS[c];
    v1S[t] = s > 0.f ? s : 0.f;
  }
  __syncthreads();
  if (t < CDIM){
    float s = bb2[t];
    for (int j=0;j<48;++j) s += w2[t*48 + j]*v1S[j];
    vbuf[t] = 1.f/(1.f + expf(-s));
  }
}

// ---- final: out = x1*skip2 + t2*v[c] ----
__global__ void k_final(const float* __restrict__ x1, const float* __restrict__ skip2,
                        const float* __restrict__ t2, const float* __restrict__ vbuf,
                        float* __restrict__ out){
  int tid = blockIdx.x*blockDim.x + threadIdx.x;
  if (tid >= LTOT*CDIM) return;
  int c = tid % CDIM;
  out[tid] = x1[tid]*skip2[c] + t2[tid]*vbuf[c];
}

static inline size_t align256(size_t x){ return (x + 255) & ~(size_t)255; }

extern "C" void kernel_launch(void* const* d_in, const int* in_sizes, int n_in,
                              void* d_out, int out_size, void* d_ws, size_t ws_size,
                              hipStream_t stream) {
  (void)in_sizes; (void)n_in; (void)out_size; (void)ws_size;
  const float* inp        = (const float*)d_in[0];
  // d_in[1] = x_size (ints) - dims are compile-time constants here
  const float* ln1_w      = (const float*)d_in[2];
  const float* ln1_b      = (const float*)d_in[3];
  const float* skip1      = (const float*)d_in[4];
  const float* skip2      = (const float*)d_in[5];
  const float* ln2_w      = (const float*)d_in[6];
  const float* ln2_b      = (const float*)d_in[7];
  const float* in_proj_w  = (const float*)d_in[8];
  const float* conv_w     = (const float*)d_in[9];
  const float* conv_b     = (const float*)d_in[10];
  const float* x_proj_w   = (const float*)d_in[11];
  const float* dt_w       = (const float*)d_in[12];
  const float* dt_b       = (const float*)d_in[13];
  const float* A_logs     = (const float*)d_in[14];
  const float* Ds         = (const float*)d_in[15];
  const float* out_norm_w = (const float*)d_in[16];
  const float* out_norm_b = (const float*)d_in[17];
  const float* out_proj_w = (const float*)d_in[18];
  const float* cab_w1     = (const float*)d_in[19];
  const float* cab_b1     = (const float*)d_in[20];
  const float* cab_w2     = (const float*)d_in[21];
  const float* cab_b2     = (const float*)d_in[22];
  const float* ca_w1      = (const float*)d_in[23];
  const float* ca_b1      = (const float*)d_in[24];
  const float* ca_w2      = (const float*)d_in[25];
  const float* ca_b2      = (const float*)d_in[26];
  float* outp = (float*)d_out;

  // ---- workspace carve-up ----
  char* ws = (char*)d_ws;
  size_t off = 0;
  auto carve = [&](size_t bytes)->char* { char* p = ws + off; off = align256(off + bytes); return p; };

  _Float16* xn_h   = (_Float16*)carve((size_t)LTOT*CDIM*2);
  float*    xf     = (float*)   carve((size_t)LTOT*DINC*4);
  float*    zf     = (float*)   carve((size_t)LTOT*DINC*4);
  float*    xc_f   = (float*)   carve((size_t)DINC*LTOT*4);
  _Float16* xc_h   = (_Float16*)carve((size_t)LTOT*DINC*2);
  float*    xdbl   = (float*)   carve((size_t)KDIR*EDIM*LTOT*4);
  float*    accb   = (float*)   carve((size_t)DINC*LTOT*4);
  _Float16* y_h    = (_Float16*)carve((size_t)LTOT*DINC*2);
  float*    x1     = (float*)   carve((size_t)LTOT*CDIM*4);
  _Float16* y2_h   = (_Float16*)carve((size_t)LTOT*CDIM*2);
  _Float16* t1_h   = (_Float16*)carve((size_t)LTOT*32*2);
  float*    t2     = (float*)   carve((size_t)LTOT*CDIM*4);
  float*    chan   = (float*)   carve(96*4);
  float*    vbuf   = (float*)   carve(96*4);
  int*      vmap   = (int*)     carve((size_t)KDIR*LTOT*4);
  _Float16* win_h  = (_Float16*)carve((size_t)192*96*2);
  _Float16* wx_h   = (_Float16*)carve((size_t)KDIR*EDIM*96*2);
  _Float16* wout_h = (_Float16*)carve((size_t)96*96*2);
  _Float16* w1t_h  = (_Float16*)carve((size_t)32*27*96*2);
  _Float16* w2t_h  = (_Float16*)carve((size_t)96*27*32*2);

  // ---- weight conversions & index map ----
  k_f2h<<<(192*96 + 255)/256, 256, 0, stream>>>(in_proj_w, win_h, 192*96);
  k_f2h<<<(KDIR*EDIM*96 + 255)/256, 256, 0, stream>>>(x_proj_w, wx_h, KDIR*EDIM*96);
  k_f2h<<<(96*96 + 255)/256, 256, 0, stream>>>(out_proj_w, wout_h, 96*96);
  k_tw1<<<(32*27*96 + 255)/256, 256, 0, stream>>>(cab_w1, w1t_h);
  k_tw2<<<(96*27*32 + 255)/256, 256, 0, stream>>>(cab_w2, w2t_h);
  k_vmap<<<(KDIR*LTOT + 255)/256, 256, 0, stream>>>(vmap);

  // ---- pipeline ----
  k_ln1<<<LTOT/256, 256, 0, stream>>>(inp, ln1_w, ln1_b, xn_h);
  k_inproj<<<dim3(LTOT/16, 3), 32, 0, stream>>>(xn_h, win_h, xf, zf);
  k_dwconv<<<(LTOT*DINC)/256, 256, 0, stream>>>(xf, conv_w, conv_b, xc_f, xc_h);
  k_xproj<<<dim3(LTOT/16, KDIR), 32, 0, stream>>>(xc_h, wx_h, vmap, xdbl);

  (void)hipMemsetAsync(accb, 0, (size_t)DINC*LTOT*4, stream);
  k_scan<<<(KDIR*DINC)/8, 128, 0, stream>>>(xdbl, xc_f, vmap, dt_w, dt_b, A_logs, Ds, accb);

  k_outnorm<<<LTOT/256, 256, 0, stream>>>(accb, zf, out_norm_w, out_norm_b, y_h);
  k_outproj<<<LTOT/16, 32, 0, stream>>>(y_h, wout_h, inp, skip1, x1);
  k_ln2<<<LTOT/256, 256, 0, stream>>>(x1, ln2_w, ln2_b, y2_h);

  k_cab1<<<LTOT/16, 32, 0, stream>>>(y2_h, w1t_h, cab_b1, t1_h);
  (void)hipMemsetAsync(chan, 0, 96*4, stream);
  k_cab2<<<LTOT/16, 32, 0, stream>>>(t1_h, w2t_h, cab_b2, t2, chan);
  k_ca<<<1, 128, 0, stream>>>(chan, ca_w1, ca_b1, ca_w2, ca_b2, vbuf);
  k_final<<<(LTOT*CDIM)/256, 256, 0, stream>>>(x1, skip2, t2, vbuf, outp);
}
